// Speller_43293270344320
// MI455X (gfx1250) — matrix operand order repaired and checked by reading, not compiled, not run on hardware
//
#include <hip/hip_runtime.h>

// ---------------------------------------------------------------------------
// LAS speller decoder for MI455X (gfx1250, wave32, WMMA).
//   V=33, D_IN=512, H=256, Q=128, L=1500, N=64, T=250
// Heavy GEMMs use v_wmma_f32_16x16x32_bf16 (bf16 in, f32 accumulate).
// ---------------------------------------------------------------------------

typedef __bf16 bf16;
typedef __attribute__((ext_vector_type(16))) __bf16 bf16x16;
typedef __attribute__((ext_vector_type(8)))  float  floatx8;

#define DEV __device__ __forceinline__

constexpr int kV = 33, kD = 512, kH = 256, kQ = 128;
constexpr int kL = 1500, kN = 64, kT = 250;

DEV floatx8 fzero8() {
  floatx8 v;
#pragma unroll
  for (int i = 0; i < 8; ++i) v[i] = 0.f;
  return v;
}

DEV floatx8 wmma_bf16(bf16x16 a, bf16x16 b, floatx8 c) {
  // D = A(16x32 bf16) * B(32x16 bf16) + C(16x16 f32)
  return __builtin_amdgcn_wmma_f32_16x16x32_bf16(false, a, false, b,
                                                 (short)0, c, false, false);
}

// A fragment: 16x32 tile of a row-major (M x K) bf16 matrix.
// Lane layout (ISA 7.12.2): lanes 0-15 M=0..15; elems 0-7 -> K=k0+8*half+e,
// elems 8-15 -> K=k0+16+8*half+(e-8).
DEV bf16x16 load_a_bf16(const bf16* A, int lda, int m0, int k0, int lane) {
  const bf16* p = A + (m0 + (lane & 15)) * lda + k0 + ((lane >> 4) << 3);
  bf16x16 a;
#pragma unroll
  for (int e = 0; e < 8; ++e) { a[e] = p[e]; a[e + 8] = p[e + 16]; }
  return a;
}

DEV bf16x16 load_a_f32(const float* A, int lda, int m0, int k0, int lane) {
  const float* p = A + (long)(m0 + (lane & 15)) * lda + k0 + ((lane >> 4) << 3);
  bf16x16 a;
#pragma unroll
  for (int e = 0; e < 8; ++e) { a[e] = (bf16)p[e]; a[e + 8] = (bf16)p[e + 16]; }
  return a;
}

// B fragment: 32x16 (K x N) tile, sourced from a row-major (N x K) weight
// (i.e. B = W^T). elems 0-7 -> K = k0 + 8*half + e, elems 8-15 ->
// K = k0 + 16 + 8*half + (e-8), N = n0 + (lane&15).
DEV bf16x16 load_b_nk(const bf16* W, int ldw, int n0, int k0, int lane) {
  const bf16* p = W + (n0 + (lane & 15)) * ldw + k0 + ((lane >> 4) << 3);
  bf16x16 b;
#pragma unroll
  for (int e = 0; e < 8; ++e) { b[e] = p[e]; b[e + 8] = p[e + 16]; }
  return b;
}

DEV float sigm(float x) { return 1.f / (1.f + expf(-x)); }
DEV float eluf(float x) { return x > 0.f ? x : expm1f(x); }

// ---------------------------------------------------------------------------
// Weight / bias preparation kernels
// ---------------------------------------------------------------------------
__global__ __launch_bounds__(256) void k_cvt(const float* __restrict__ s,
                                             bf16* __restrict__ d, int n) {
  int i = blockIdx.x * blockDim.x + threadIdx.x;
  if (i < n) d[i] = (bf16)s[i];
}

// dst row r = [A row r (ka cols) | B row r (kb cols)]  -> bf16
__global__ __launch_bounds__(256) void k_cat2(const float* __restrict__ A, int ka,
                                              const float* __restrict__ B, int kb,
                                              bf16* __restrict__ dst, int rows) {
  int K = ka + kb;
  int i = blockIdx.x * blockDim.x + threadIdx.x;
  if (i >= rows * K) return;
  int r = i / K, k = i - r * K;
  float v = (k < ka) ? A[r * ka + k] : B[r * kb + (k - ka)];
  dst[i] = (bf16)v;
}

__global__ __launch_bounds__(256) void k_addb(const float* __restrict__ a,
                                              const float* __restrict__ b,
                                              float* __restrict__ d, int n) {
  int i = blockIdx.x * blockDim.x + threadIdx.x;
  if (i < n) d[i] = a[i] + b[i];
}

// ---------------------------------------------------------------------------
// Encoder: fused  out = (ELU(X @ W1^T + b1)) @ W2^T + b2  per 32-row tile.
// X = seqs (L*N x 512) fp32, hidden staged in LDS as bf16.
// MODE 0: key features, scatter (N, Q, L).  MODE 1: value features, (N, L, H).
// ---------------------------------------------------------------------------
template <int N1, int N2, int MODE>
__global__ __launch_bounds__(256) void encoder_kernel(
    const float* __restrict__ X, const bf16* __restrict__ W1,
    const float* __restrict__ b1, const bf16* __restrict__ W2,
    const float* __restrict__ b2, bf16* __restrict__ out) {
  __shared__ bf16 hid[32 * N1];
  const int tid = threadIdx.x, lane = tid & 31, wid = tid >> 5;
  const int col0 = lane & 15, hl = lane >> 4;
  const long row0 = (long)blockIdx.x * 32;

  {  // phase 1: hid = ELU(X @ W1^T + b1), 32 x N1
    constexpr int TN = N1 / 16, TPW = TN / 4;
    const int mt = wid >> 2, grp = wid & 3;
    floatx8 acc[TPW];
#pragma unroll
    for (int j = 0; j < TPW; ++j) acc[j] = fzero8();
    const float* Ab = X + (row0 + mt * 16) * kD;
    for (int k0 = 0; k0 < kD; k0 += 32) {
      bf16x16 a = load_a_f32(Ab, kD, 0, k0, lane);
#pragma unroll
      for (int j = 0; j < TPW; ++j) {
        bf16x16 b = load_b_nk(W1, kD, (grp * TPW + j) * 16, k0, lane);
        acc[j] = wmma_bf16(a, b, acc[j]);
      }
    }
#pragma unroll
    for (int j = 0; j < TPW; ++j) {
      int col = (grp * TPW + j) * 16 + col0;
      float bv = b1[col];
#pragma unroll
      for (int r = 0; r < 8; ++r) {
        int m = mt * 16 + r + 8 * hl;
        hid[m * N1 + col] = (bf16)eluf(acc[j][r] + bv);
      }
    }
  }
  __syncthreads();
  {  // phase 2: out = hid @ W2^T + b2, 32 x N2, scatter to feature layout
    constexpr int TN2 = N2 / 16, TPW2 = TN2 / 4;
    const int mt = wid >> 2, grp = wid & 3;
    floatx8 acc[TPW2];
#pragma unroll
    for (int j = 0; j < TPW2; ++j) acc[j] = fzero8();
    for (int k0 = 0; k0 < N1; k0 += 32) {
      bf16x16 a = load_a_bf16(hid, N1, mt * 16, k0, lane);
#pragma unroll
      for (int j = 0; j < TPW2; ++j) {
        bf16x16 b = load_b_nk(W2, N1, (grp * TPW2 + j) * 16, k0, lane);
        acc[j] = wmma_bf16(a, b, acc[j]);
      }
    }
#pragma unroll
    for (int j = 0; j < TPW2; ++j) {
      int col = (grp * TPW2 + j) * 16 + col0;
      float bv = b2[col];
#pragma unroll
      for (int r = 0; r < 8; ++r) {
        long grow = row0 + mt * 16 + r + 8 * hl;  // row in (L*N)
        int l = (int)(grow >> 6), n = (int)(grow & 63);
        float v = acc[j][r] + bv;
        if (MODE == 0)
          out[((long)n * kQ + col) * kL + l] = (bf16)v;   // (N, Q, L)
        else
          out[((long)n * kL + l) * kH + col] = (bf16)v;   // (N, L, H)
      }
    }
  }
}

// ---------------------------------------------------------------------------
// Attention: one block per batch element. scores -> softmax -> mask/clamp ->
// renormalize -> context. Key/value features stream from L2 (74MB working set).
// ---------------------------------------------------------------------------
__global__ __launch_bounds__(256) void attn_kernel(
    const bf16* __restrict__ kf, const bf16* __restrict__ vf,
    const float* __restrict__ q, const int* __restrict__ seq_lens,
    bf16* __restrict__ X0) {
  __shared__ float sc[kL];
  __shared__ float qs[kQ];
  __shared__ float red[256];
  const int n = blockIdx.x, tid = threadIdx.x;

  if (tid < kQ) qs[tid] = q[n * kQ + tid];
  __syncthreads();

  // scores[l] = sum_q kf[n][q][l] * qs[q]  (coalesced over l)
  const bf16* kp = kf + (long)n * kQ * kL;
  for (int l = tid; l < kL; l += 256) {
    float a = 0.f;
#pragma unroll 4
    for (int qi = 0; qi < kQ; ++qi) a += (float)kp[qi * kL + l] * qs[qi];
    sc[l] = a;
  }
  __syncthreads();

  // block max
  float lm = -3.0e38f;
  for (int l = tid; l < kL; l += 256) lm = fmaxf(lm, sc[l]);
  red[tid] = lm;
  __syncthreads();
  for (int s = 128; s > 0; s >>= 1) {
    if (tid < s) red[tid] = fmaxf(red[tid], red[tid + s]);
    __syncthreads();
  }
  const float mx = red[0];
  __syncthreads();

  // exp and sum
  float ls = 0.f;
  for (int l = tid; l < kL; l += 256) {
    float e = expf(sc[l] - mx);
    sc[l] = e;
    ls += e;
  }
  red[tid] = ls;
  __syncthreads();
  for (int s = 128; s > 0; s >>= 1) {
    if (tid < s) red[tid] += red[tid + s];
    __syncthreads();
  }
  const float E = red[0];  // softmax denominator
  __syncthreads();

  // a = e/E ; b = max(a*mask, 1e-9) ; b /= sum(b)
  // equivalently w = max(mask? e : 0, 1e-9*E), normalized at the end.
  const float thr = 1e-9f * E;
  const int sl = seq_lens[n];
  float s2 = 0.f;
  for (int l = tid; l < kL; l += 256) {
    float wv = fmaxf((l < sl) ? sc[l] : 0.f, thr);
    sc[l] = wv;
    s2 += wv;
  }
  red[tid] = s2;
  __syncthreads();
  for (int s = 128; s > 0; s >>= 1) {
    if (tid < s) red[tid] += red[tid + s];
    __syncthreads();
  }
  const float inv = 1.f / red[0];

  // context: thread tid owns channel h = tid (coalesced over h)
  const bf16* vp = vf + (long)n * kL * kH + tid;
  float acc = 0.f;
  for (int l = 0; l < kL; ++l) acc += sc[l] * (float)vp[l * kH];
  X0[n * 768 + tid] = (bf16)(acc * inv);  // ctx slot of X0 = [ctx|emb|h0prev]
}

// ---------------------------------------------------------------------------
// Decoder step: single 1024-thread (32 wave) workgroup; phases separated by
// workgroup barriers. All matmuls via WMMA bf16.
// ---------------------------------------------------------------------------
struct DecArgs {
  const float *W_emb, *inith, *bq1, *bq2, *bo1, *bo2;
  const int* label_in;
  const float *b0, *b1, *b2;                    // combined LSTM biases (f32)
  const bf16 *Wc0, *Wc1, *Wc2, *Wq1, *Wq2, *Wo1;  // bf16 weights (N x K)
  bf16 *X0, *X1, *X2, *Xq, *Xqh;                // activation concat buffers
  float *g, *c0, *c1, *c2, *q, *l1, *out;
};

// g = X(64 x K) @ W^T(1024 x K) + bias -> g (64 x 1024) f32
DEV void gemm_gates(const bf16* __restrict__ X, int ldx,
                    const bf16* __restrict__ W, int K,
                    const float* __restrict__ bias, float* __restrict__ gout,
                    int lane, int wid) {
  const int col0 = lane & 15, hl = lane >> 4;
  const int mt = wid >> 3, nt0 = (wid & 7) * 8;  // 4 m-tiles x 64 n-tiles
  floatx8 acc[8];
#pragma unroll
  for (int j = 0; j < 8; ++j) acc[j] = fzero8();
  for (int k0 = 0; k0 < K; k0 += 32) {
    bf16x16 a = load_a_bf16(X, ldx, mt * 16, k0, lane);
#pragma unroll
    for (int j = 0; j < 8; ++j) {
      bf16x16 b = load_b_nk(W, K, (nt0 + j) * 16, k0, lane);
      acc[j] = wmma_bf16(a, b, acc[j]);
    }
  }
#pragma unroll
  for (int j = 0; j < 8; ++j) {
    int col = (nt0 + j) * 16 + col0;
    float bv = bias[col];
#pragma unroll
    for (int r = 0; r < 8; ++r)
      gout[(mt * 16 + r + 8 * hl) * 1024 + col] = acc[j][r] + bv;
  }
}

// mode 0: init states + query ; mode 1: full step t ; mode 2: final logits only
__global__ __launch_bounds__(1024) void decoder_kernel(DecArgs A, int mode, int t) {
  const int tid = threadIdx.x, lane = tid & 31, wid = tid >> 5;
  const int col0 = lane & 15, hl = lane >> 4;

  if (mode == 0) {  // initialize recurrent state from inith, c = 0
    for (int idx = tid; idx < kN * kH; idx += 1024) {
      int n = idx >> 8, h = idx & 255;
      A.c0[idx] = 0.f; A.c1[idx] = 0.f; A.c2[idx] = 0.f;
      A.X0[n * 768 + 512 + h] = (bf16)A.inith[h];
      A.X1[n * 512 + 256 + h] = (bf16)A.inith[256 + h];
      bf16 h2 = (bf16)A.inith[512 + h];
      A.X2[n * 512 + 256 + h] = h2;
      A.Xq[n * 256 + h] = h2;
    }
    __syncthreads();
  }

  if (mode >= 1 && t > 0) {
    // P0: logits_{t-1} = ([ctx|h2] @ Wo1^T + bo1) @ W_emb^T + bo2
    {
      const int mt = wid >> 3, nt0 = (wid & 7) * 2;  // 4 x 16 tiles
      floatx8 acc[2] = {fzero8(), fzero8()};
      for (int k0 = 0; k0 < 512; k0 += 32) {
        const bf16* src; int ldx, kk;
        if (k0 < 256) { src = A.X0;       ldx = 768; kk = k0; }        // ctx
        else          { src = A.X2 + 256; ldx = 512; kk = k0 - 256; }  // h2
        bf16x16 a = load_a_bf16(src, ldx, mt * 16, kk, lane);
#pragma unroll
        for (int j = 0; j < 2; ++j) {
          bf16x16 b = load_b_nk(A.Wo1, 512, (nt0 + j) * 16, k0, lane);
          acc[j] = wmma_bf16(a, b, acc[j]);
        }
      }
#pragma unroll
      for (int j = 0; j < 2; ++j) {
        int col = (nt0 + j) * 16 + col0;
        float bv = A.bo1[col];
#pragma unroll
        for (int r = 0; r < 8; ++r)
          A.l1[(mt * 16 + r + 8 * hl) * 256 + col] = acc[j][r] + bv;
      }
    }
    __syncthreads();
    for (int idx = tid; idx < kN * kV; idx += 1024) {  // tied output proj
      int n = idx / kV, v = idx - n * kV;
      const float* lr = A.l1 + n * 256;
      const float* er = A.W_emb + v * 256;
      float s = A.bo2[v];
      for (int h = 0; h < 256; ++h) s += lr[h] * er[h];
      A.out[(long)(t - 1) * (kN * kV) + idx] = s;
    }
    __syncthreads();
  }

  if (mode == 1) {
    // P1: embedding gather into X0[:, 256:512]
    for (int idx = tid; idx < kN * kH; idx += 1024) {
      int n = idx >> 8, h = idx & 255;
      int lab = A.label_in[n * kT + t];
      A.X0[n * 768 + 256 + h] = (bf16)A.W_emb[lab * 256 + h];
    }
    __syncthreads();

    // LSTM 0
    gemm_gates(A.X0, 768, A.Wc0, 768, A.b0, A.g, lane, wid);
    __syncthreads();
    for (int idx = tid; idx < kN * kH; idx += 1024) {
      int n = idx >> 8, h = idx & 255;
      const float* gr = A.g + n * 1024;
      float c = sigm(gr[256 + h]) * A.c0[idx] + sigm(gr[h]) * tanhf(gr[512 + h]);
      A.c0[idx] = c;
      float hh = sigm(gr[768 + h]) * tanhf(c);
      A.X1[n * 512 + h] = (bf16)hh;
      A.X0[n * 768 + 512 + h] = (bf16)hh;  // h0_prev for next step
    }
    __syncthreads();

    // LSTM 1
    gemm_gates(A.X1, 512, A.Wc1, 512, A.b1, A.g, lane, wid);
    __syncthreads();
    for (int idx = tid; idx < kN * kH; idx += 1024) {
      int n = idx >> 8, h = idx & 255;
      const float* gr = A.g + n * 1024;
      float c = sigm(gr[256 + h]) * A.c1[idx] + sigm(gr[h]) * tanhf(gr[512 + h]);
      A.c1[idx] = c;
      float hh = sigm(gr[768 + h]) * tanhf(c);
      A.X2[n * 512 + h] = (bf16)hh;
      A.X1[n * 512 + 256 + h] = (bf16)hh;  // h1_prev for next step
    }
    __syncthreads();

    // LSTM 2
    gemm_gates(A.X2, 512, A.Wc2, 512, A.b2, A.g, lane, wid);
    __syncthreads();
    for (int idx = tid; idx < kN * kH; idx += 1024) {
      int n = idx >> 8, h = idx & 255;
      const float* gr = A.g + n * 1024;
      float c = sigm(gr[256 + h]) * A.c2[idx] + sigm(gr[h]) * tanhf(gr[512 + h]);
      A.c2[idx] = c;
      float hh = sigm(gr[768 + h]) * tanhf(c);
      A.Xq[n * 256 + h] = (bf16)hh;
      A.X2[n * 512 + 256 + h] = (bf16)hh;  // h2_prev for next step
    }
    __syncthreads();
  }

  if (mode <= 1) {
    // P8: Xqh = ELU(h2 @ Wq1^T + bq1)
    {
      const int mt = wid >> 3, nt0 = (wid & 7) * 2;  // 4 x 16 tiles
      floatx8 acc[2] = {fzero8(), fzero8()};
      for (int k0 = 0; k0 < 256; k0 += 32) {
        bf16x16 a = load_a_bf16(A.Xq, 256, mt * 16, k0, lane);
#pragma unroll
        for (int j = 0; j < 2; ++j) {
          bf16x16 b = load_b_nk(A.Wq1, 256, (nt0 + j) * 16, k0, lane);
          acc[j] = wmma_bf16(a, b, acc[j]);
        }
      }
#pragma unroll
      for (int j = 0; j < 2; ++j) {
        int col = (nt0 + j) * 16 + col0;
        float bv = A.bq1[col];
#pragma unroll
        for (int r = 0; r < 8; ++r)
          A.Xqh[(mt * 16 + r + 8 * hl) * 256 + col] = (bf16)eluf(acc[j][r] + bv);
      }
    }
    __syncthreads();
    // P9: q = Xqh @ Wq2^T + bq2  (64 x 128 f32)
    {
      const int mt = wid >> 3, nt = wid & 7;  // 4 x 8 tiles, one per wave
      floatx8 acc = fzero8();
      for (int k0 = 0; k0 < 256; k0 += 32) {
        bf16x16 a = load_a_bf16(A.Xqh, 256, mt * 16, k0, lane);
        bf16x16 b = load_b_nk(A.Wq2, 256, nt * 16, k0, lane);
        acc = wmma_bf16(a, b, acc);
      }
      int col = nt * 16 + col0;
      float bv = A.bq2[col];
#pragma unroll
      for (int r = 0; r < 8; ++r)
        A.q[(mt * 16 + r + 8 * hl) * kQ + col] = acc[r] + bv;
    }
  }
}

// ---------------------------------------------------------------------------
// Host side
// ---------------------------------------------------------------------------
extern "C" void kernel_launch(void* const* d_in, const int* in_sizes, int n_in,
                              void* d_out, int out_size, void* d_ws, size_t ws_size,
                              hipStream_t stream) {
  const float* seqs     = (const float*)d_in[0];
  const int*   seq_lens = (const int*)d_in[1];
  const int*   label_in = (const int*)d_in[2];
  const float* W_emb    = (const float*)d_in[3];
  const float* inith    = (const float*)d_in[4];
  const float* Wih0 = (const float*)d_in[5];
  const float* Whh0 = (const float*)d_in[6];
  const float* bih0 = (const float*)d_in[7];
  const float* bhh0 = (const float*)d_in[8];
  const float* Wih1 = (const float*)d_in[9];
  const float* Whh1 = (const float*)d_in[10];
  const float* bih1 = (const float*)d_in[11];
  const float* bhh1 = (const float*)d_in[12];
  const float* Wih2 = (const float*)d_in[13];
  const float* Whh2 = (const float*)d_in[14];
  const float* bih2 = (const float*)d_in[15];
  const float* bhh2 = (const float*)d_in[16];
  const float* Wq1 = (const float*)d_in[17];
  const float* bq1 = (const float*)d_in[18];
  const float* Wq2 = (const float*)d_in[19];
  const float* bq2 = (const float*)d_in[20];
  const float* Wk1 = (const float*)d_in[21];
  const float* bk1 = (const float*)d_in[22];
  const float* Wk2 = (const float*)d_in[23];
  const float* bk2 = (const float*)d_in[24];
  const float* Wv1 = (const float*)d_in[25];
  const float* bv1 = (const float*)d_in[26];
  const float* Wv2 = (const float*)d_in[27];
  const float* bv2 = (const float*)d_in[28];
  const float* Wo1 = (const float*)d_in[29];
  const float* bo1 = (const float*)d_in[30];
  const float* bo2 = (const float*)d_in[31];
  float* out = (float*)d_out;

  char* wp = (char*)d_ws;
  auto alloc = [&](size_t bytes) -> void* {
    void* r = (void*)wp;
    wp += (bytes + 255) & ~(size_t)255;
    return r;
  };

  bf16* Wc0  = (bf16*)alloc((size_t)1024 * 768 * 2);
  bf16* Wc1  = (bf16*)alloc((size_t)1024 * 512 * 2);
  bf16* Wc2  = (bf16*)alloc((size_t)1024 * 512 * 2);
  bf16* Wq1b = (bf16*)alloc((size_t)256 * 256 * 2);
  bf16* Wq2b = (bf16*)alloc((size_t)128 * 256 * 2);
  bf16* Wk1b = (bf16*)alloc((size_t)512 * 512 * 2);
  bf16* Wk2b = (bf16*)alloc((size_t)128 * 512 * 2);
  bf16* Wv1b = (bf16*)alloc((size_t)256 * 512 * 2);
  bf16* Wv2b = (bf16*)alloc((size_t)256 * 256 * 2);
  bf16* Wo1b = (bf16*)alloc((size_t)256 * 512 * 2);
  float* b0c = (float*)alloc(1024 * 4);
  float* b1c = (float*)alloc(1024 * 4);
  float* b2c = (float*)alloc(1024 * 4);
  bf16* kf = (bf16*)alloc((size_t)kN * kQ * kL * 2);  // (N,Q,L)
  bf16* vf = (bf16*)alloc((size_t)kN * kL * kH * 2);  // (N,L,H)
  bf16* X0 = (bf16*)alloc((size_t)kN * 768 * 2);      // [ctx|emb|h0prev]
  bf16* X1 = (bf16*)alloc((size_t)kN * 512 * 2);      // [h0|h1prev]
  bf16* X2 = (bf16*)alloc((size_t)kN * 512 * 2);      // [h1|h2prev]
  bf16* Xq  = (bf16*)alloc((size_t)kN * 256 * 2);     // h2
  bf16* Xqh = (bf16*)alloc((size_t)kN * 256 * 2);     // ELU(h2 Wq1)
  float* gbuf = (float*)alloc((size_t)kN * 1024 * 4);
  float* c0 = (float*)alloc((size_t)kN * kH * 4);
  float* c1 = (float*)alloc((size_t)kN * kH * 4);
  float* c2 = (float*)alloc((size_t)kN * kH * 4);
  float* qb = (float*)alloc((size_t)kN * kQ * 4);
  float* l1 = (float*)alloc((size_t)kN * kH * 4);

  // --- weight preparation (bf16 packing, [Wih|Whh] concat, bias sums) ---
  auto g1 = [](int n) { return dim3((n + 255) / 256); };
  k_cat2<<<g1(1024 * 768), 256, 0, stream>>>(Wih0, 512, Whh0, 256, Wc0, 1024);
  k_cat2<<<g1(1024 * 512), 256, 0, stream>>>(Wih1, 256, Whh1, 256, Wc1, 1024);
  k_cat2<<<g1(1024 * 512), 256, 0, stream>>>(Wih2, 256, Whh2, 256, Wc2, 1024);
  k_cvt<<<g1(256 * 256), 256, 0, stream>>>(Wq1, Wq1b, 256 * 256);
  k_cvt<<<g1(128 * 256), 256, 0, stream>>>(Wq2, Wq2b, 128 * 256);
  k_cvt<<<g1(512 * 512), 256, 0, stream>>>(Wk1, Wk1b, 512 * 512);
  k_cvt<<<g1(128 * 512), 256, 0, stream>>>(Wk2, Wk2b, 128 * 512);
  k_cvt<<<g1(256 * 512), 256, 0, stream>>>(Wv1, Wv1b, 256 * 512);
  k_cvt<<<g1(256 * 256), 256, 0, stream>>>(Wv2, Wv2b, 256 * 256);
  k_cvt<<<g1(256 * 512), 256, 0, stream>>>(Wo1, Wo1b, 256 * 512);
  k_addb<<<g1(1024), 256, 0, stream>>>(bih0, bhh0, b0c, 1024);
  k_addb<<<g1(1024), 256, 0, stream>>>(bih1, bhh1, b1c, 1024);
  k_addb<<<g1(1024), 256, 0, stream>>>(bih2, bhh2, b2c, 1024);

  // --- encoder feature GEMMs (fused MLP2 through LDS) ---
  const int nblk = (kL * kN) / 32;  // 3000
  encoder_kernel<512, 128, 0><<<nblk, 256, 0, stream>>>(seqs, Wk1b, bk1, Wk2b, bk2, kf);
  encoder_kernel<256, 256, 1><<<nblk, 256, 0, stream>>>(seqs, Wv1b, bv1, Wv2b, bv2, vf);

  DecArgs DA;
  DA.W_emb = W_emb; DA.inith = inith; DA.bq1 = bq1; DA.bq2 = bq2;
  DA.bo1 = bo1; DA.bo2 = bo2; DA.label_in = label_in;
  DA.b0 = b0c; DA.b1 = b1c; DA.b2 = b2c;
  DA.Wc0 = Wc0; DA.Wc1 = Wc1; DA.Wc2 = Wc2;
  DA.Wq1 = Wq1b; DA.Wq2 = Wq2b; DA.Wo1 = Wo1b;
  DA.X0 = X0; DA.X1 = X1; DA.X2 = X2; DA.Xq = Xq; DA.Xqh = Xqh;
  DA.g = gbuf; DA.c0 = c0; DA.c1 = c1; DA.c2 = c2;
  DA.q = qb; DA.l1 = l1; DA.out = out;

  // --- decode loop: init query/attention, then T steps, then final logits ---
  decoder_kernel<<<1, 1024, 0, stream>>>(DA, 0, 0);
  attn_kernel<<<kN, 256, 0, stream>>>(kf, vf, qb, seq_lens, X0);
  for (int t = 0; t < kT; ++t) {
    decoder_kernel<<<1, 1024, 0, stream>>>(DA, 1, t);
    attn_kernel<<<kN, 256, 0, stream>>>(kf, vf, qb, seq_lens, X0);
  }
  decoder_kernel<<<1, 1024, 0, stream>>>(DA, 2, kT);
  (void)in_sizes; (void)n_in; (void)out_size; (void)ws_size;
}
